// CustomLSTMModel_23089744183525
// MI455X (gfx1250) — matrix operand order repaired and checked by reading, not compiled, not run on hardware
//
#include <hip/hip_runtime.h>
#include <hip/hip_bf16.h>

typedef __attribute__((ext_vector_type(16))) __bf16 bfx16;
typedef __attribute__((ext_vector_type(8)))  __bf16 bfx8;
typedef __attribute__((ext_vector_type(8)))  float  fx8;

#define HID 512
#define EMB 512
#define KDIM 1024
#define BATCH 64
#define SEQ 512
#define NLAYER 2
#define RS 1032                      // padded LDS row stride (bf16 elems): 516 DWORDs
#define LDS_BYTES ((64 + 64) * RS * 2)  // 64 A-rows (4 gates x 16) + 64 B-cols = 264192

__device__ __forceinline__ float sigmoid_(float x) {
    return 1.0f / (1.0f + __expf(-x));
}
__device__ __forceinline__ float tanh_(float x) {
    float cl = fminf(fmaxf(x, -15.0f), 15.0f);
    float e  = __expf(2.0f * cl);
    return (e - 1.0f) / (e + 1.0f);
}

// Load a 16-element bf16 fragment from two 16B-aligned 8-element halves `gap` apart.
__device__ __forceinline__ bfx16 load_frag16(const __bf16* p, int gap) {
    bfx8 lo = *(const bfx8*)(p);
    bfx8 hi = *(const bfx8*)(p + gap);
    return __builtin_shufflevector(lo, hi, 0, 1, 2, 3, 4, 5, 6, 7, 8, 9, 10, 11, 12,
                                   13, 14, 15);
}

#define WMMA_BF16(a, b, c)                                                            \
    __builtin_amdgcn_wmma_f32_16x16x32_bf16(false, a, false, b, (short)0, c, false,   \
                                            false)

// ---------------- prep: fp32 weights -> packed bf16 [layer][gate f,i,c,o][H][K] ----
__global__ void convert_weights(const float* __restrict__ Wf, const float* __restrict__ Wi,
                                const float* __restrict__ Wc, const float* __restrict__ Wo,
                                __bf16* __restrict__ Wb) {
    const size_t HK  = (size_t)HID * KDIM;      // per layer per gate
    const size_t tot = (size_t)NLAYER * HK;     // per gate
    for (size_t i = (size_t)blockIdx.x * blockDim.x + threadIdx.x; i < tot;
         i += (size_t)gridDim.x * blockDim.x) {
        size_t l = i / HK, r = i % HK;
        size_t d = l * 4 * HK + r;
        Wb[d + 0 * HK] = (__bf16)Wf[i];
        Wb[d + 1 * HK] = (__bf16)Wi[i];
        Wb[d + 2 * HK] = (__bf16)Wc[i];
        Wb[d + 3 * HK] = (__bf16)Wo[i];
    }
}

// ---------------- prep: gather all embeddings, transposed bf16 [t][n][e] ------------
__global__ void embed_all(const int* __restrict__ texts, const float* __restrict__ emb,
                          __bf16* __restrict__ embT) {
    const size_t tot = (size_t)SEQ * BATCH * EMB;
    for (size_t i = (size_t)blockIdx.x * blockDim.x + threadIdx.x; i < tot;
         i += (size_t)gridDim.x * blockDim.x) {
        size_t e  = i % EMB;
        size_t tn = i / EMB;                 // t*BATCH + n
        int row = texts[tn];
        embT[i] = (__bf16)emb[(size_t)row * EMB + e];
    }
}

// ---------------- prep: zero h0 (bf16, transposed) and c (fp32, transposed) ---------
__global__ void init_state(__bf16* __restrict__ h0, float* __restrict__ cT) {
    int i = blockIdx.x * blockDim.x + threadIdx.x;
    if (i < BATCH * HID) { h0[i] = (__bf16)0.0f; cT[i] = 0.0f; }
}

// ---------------- one LSTM layer-step: LDS-staged tiles + pipelined bf16 WMMA -------
// Grid: 32 blocks (one per M-tile) x 128 threads (4 waves = 4 N-tiles).
// LDS: A slab = [gate 0..3][row 0..15][K 0..1023] padded to RS; B slab = [col 0..63][K].
__global__ void __launch_bounds__(128) lstm_step(
    const __bf16* __restrict__ W4, const float* __restrict__ bF,
    const float* __restrict__ bI, const float* __restrict__ bC,
    const float* __restrict__ bO, const __bf16* __restrict__ hin,
    const __bf16* __restrict__ xin, __bf16* __restrict__ hout,
    float* __restrict__ cT) {
    extern __shared__ __bf16 lds[];
    __bf16* Ab = lds;                 // 64 rows * RS
    __bf16* Bb = lds + 64 * RS;       // 64 cols * RS

    const int t     = threadIdx.x;
    const int Mt    = blockIdx.x;     // 0..31
    const int Nt    = t >> 5;         // wave id = N-tile 0..3
    const int lane  = t & 31;
    const int lhalf = lane >> 4;
    const int l15   = lane & 15;
    const size_t GS = (size_t)HID * KDIM;

    // ---- bulk fill: A slab (4 gates x 16 rows x 1024) in 16B units ------------------
    for (int u = t; u < 8192; u += 128) {
        int g = u >> 11, row = (u >> 7) & 15, c = u & 127;
        bfx8 v = *(const bfx8*)(W4 + g * GS + (size_t)(Mt * 16 + row) * KDIM + c * 8);
        *(bfx8*)(Ab + (g * 16 + row) * RS + c * 8) = v;
    }
    // ---- bulk fill: B slab (64 cat columns: K 0..511 from h, 512..1023 from x) ------
    for (int u = t; u < 8192; u += 128) {
        int col = u >> 7, c = u & 127;
        const __bf16* src = (c < 64) ? (hin + (size_t)col * HID + c * 8)
                                     : (xin + (size_t)col * EMB + (c - 64) * 8);
        *(bfx8*)(Bb + col * RS + c * 8) = *(const bfx8*)src;
    }

    // accumulators init with bias; C/D layout: VGPR r, lane -> M = Mt*16 + r + 8*lhalf
    fx8 accF, accI, accC, accO;
#pragma unroll
    for (int r = 0; r < 8; ++r) {
        int m = Mt * 16 + r + 8 * lhalf;
        accF[r] = bF[m]; accI[r] = bI[m]; accC[r] = bC[m]; accO[r] = bO[m];
    }

    __syncthreads();

    // Per-lane LDS pointers; all loop loads use small immediate offsets.
    const __bf16* aP0 = Ab + (0 * 16 + l15) * RS + lhalf * 8;
    const __bf16* aP1 = Ab + (1 * 16 + l15) * RS + lhalf * 8;
    const __bf16* aP2 = Ab + (2 * 16 + l15) * RS + lhalf * 8;
    const __bf16* aP3 = Ab + (3 * 16 + l15) * RS + lhalf * 8;
    const __bf16* bP  = Bb + (Nt * 16 + l15) * RS + lhalf * 8;

    // ---- fully unrolled 1-chunk-ahead ping-pong pipeline ----------------------------
    bfx16 bS[2], aFS[2], aIS[2], aCS[2], aOS[2];
    bS[0]  = load_frag16(bP, 16);
    aFS[0] = load_frag16(aP0, 16);
    aIS[0] = load_frag16(aP1, 16);
    aCS[0] = load_frag16(aP2, 16);
    aOS[0] = load_frag16(aP3, 16);
#pragma unroll
    for (int kc = 0; kc < 32; ++kc) {
        const int cur = kc & 1, nxt = cur ^ 1;
        if (kc < 31) {
            const int k1 = (kc + 1) * 32;
            bS[nxt]  = load_frag16(bP + k1, 16);
            aFS[nxt] = load_frag16(aP0 + k1, 16);
            aIS[nxt] = load_frag16(aP1 + k1, 16);
            aCS[nxt] = load_frag16(aP2 + k1, 16);
            aOS[nxt] = load_frag16(aP3 + k1, 16);
        }
        __builtin_amdgcn_sched_barrier(0);  // keep chunk k+1 loads above chunk k MMAs
        accF = WMMA_BF16(aFS[cur], bS[cur], accF);
        accI = WMMA_BF16(aIS[cur], bS[cur], accI);
        accC = WMMA_BF16(aCS[cur], bS[cur], accC);
        accO = WMMA_BF16(aOS[cur], bS[cur], accO);
        __builtin_amdgcn_sched_barrier(0);
    }

    // cell update entirely in registers; lane owns (nB, m0..m0+7) contiguously
    int nB = Nt * 16 + l15;
    int m0 = Mt * 16 + 8 * lhalf;
    float*  crow = cT   + (size_t)nB * HID + m0;
    __bf16* hrow = hout + (size_t)nB * HID + m0;
    bfx8 hv;
#pragma unroll
    for (int r = 0; r < 8; ++r) {
        float f  = sigmoid_(accF[r]);
        float i  = sigmoid_(accI[r]);
        float ch = tanh_(accC[r]);
        float o  = sigmoid_(accO[r]);
        float c  = f * crow[r] + i * ch;
        crow[r]  = c;
        hv[r]    = (__bf16)(o * tanh_(c));
    }
    *(bfx8*)hrow = hv;
}

// ---------------- final projection: y[n][o] = sum_m h[n][m]*Wy[o][m] + by[o] --------
__global__ void final_proj(const __bf16* __restrict__ hT, const float* __restrict__ Wy,
                           const float* __restrict__ by, float* __restrict__ out) {
    int i = threadIdx.x;
    if (i < BATCH * 2) {
        int n = i >> 1, o = i & 1;
        float s = by[o];
        const __bf16* hc = hT + (size_t)n * HID;
        const float*  w  = Wy + (size_t)o * HID;
        for (int k = 0; k < HID; ++k) s += (float)hc[k] * w[k];
        out[n * 2 + o] = s;
    }
}

extern "C" void kernel_launch(void* const* d_in, const int* in_sizes, int n_in,
                              void* d_out, int out_size, void* d_ws, size_t ws_size,
                              hipStream_t stream) {
    (void)in_sizes; (void)n_in; (void)out_size; (void)ws_size;
    const int*   texts = (const int*)d_in[0];
    const float* emb   = (const float*)d_in[1];
    const float* Wf    = (const float*)d_in[2];
    const float* bf_   = (const float*)d_in[3];
    const float* Wi    = (const float*)d_in[4];
    const float* bi_   = (const float*)d_in[5];
    const float* Wo    = (const float*)d_in[6];
    const float* bo_   = (const float*)d_in[7];
    const float* Wc    = (const float*)d_in[8];
    const float* bc_   = (const float*)d_in[9];
    const float* Wy    = (const float*)d_in[10];
    const float* by    = (const float*)d_in[11];
    float* out = (float*)d_out;

    char* ws = (char*)d_ws;
    size_t off = 0;
    __bf16* Wb    = (__bf16*)(ws + off); off += (size_t)NLAYER * 4 * HID * KDIM * 2; // 8 MB
    __bf16* embT  = (__bf16*)(ws + off); off += (size_t)SEQ * BATCH * EMB * 2;       // 32 MB
    __bf16* hbuf0 = (__bf16*)(ws + off); off += (size_t)BATCH * HID * 2;
    __bf16* hbuf1 = (__bf16*)(ws + off); off += (size_t)BATCH * HID * 2;
    float*  cT    = (float*)(ws + off);  off += (size_t)BATCH * HID * 4;

    convert_weights<<<1024, 256, 0, stream>>>(Wf, Wi, Wc, Wo, Wb);
    embed_all<<<4096, 256, 0, stream>>>(texts, emb, embT);
    init_state<<<(BATCH * HID + 255) / 256, 256, 0, stream>>>(hbuf0, cT);

    __bf16* hb[2] = {hbuf0, hbuf1};
    for (int t = 0; t < SEQ; ++t) {
        for (int l = 0; l < NLAYER; ++l) {
            int s = t * NLAYER + l;
            const __bf16* hin  = hb[s & 1];
            __bf16*       hout = hb[(s + 1) & 1];
            const __bf16* xin  = (l == 0) ? (embT + (size_t)t * BATCH * EMB) : hin;
            lstm_step<<<32, 128, LDS_BYTES, stream>>>(
                Wb + (size_t)l * 4 * HID * KDIM,
                bf_ + l * HID, bi_ + l * HID, bc_ + l * HID, bo_ + l * HID,
                hin, xin, hout, cT);
        }
    }
    final_proj<<<1, 128, 0, stream>>>(hb[0], Wy, by, out);
}
